// Net_36601711296722
// MI455X (gfx1250) — matrix operand order repaired and checked by reading, not compiled, not run on hardware
//
#include <hip/hip_runtime.h>
#include <stdint.h>

typedef __attribute__((ext_vector_type(16))) _Float16 v16h;
typedef __attribute__((ext_vector_type(8)))  _Float16 v8h;
typedef __attribute__((ext_vector_type(8)))  float    v8f;

#define NUM_STEPS 25
#define IN_F  20
#define HID   21
#define OUT_F 200
#define KPAD  32                 // padded K for both layers (f16 WMMA K=32)
#define H_TILES 2                // ceil(21/16)
#define O_TILES 13               // ceil(200/16)
#define HPAD (H_TILES * 16)      // 32
#define OPAD (O_TILES * 16)      // 208
#define WAVES_PER_BLOCK 4
#define BLOCK (WAVES_PER_BLOCK * 32)

#define BETA_H 0.8f
#define BETA_O 0.4f
#define THR_H  0.45f
#define THR_O  0.75f

// counter-based hash -> uniform [0,1); deterministic per (t,row,k)
__device__ __forceinline__ uint32_t hash3(uint32_t a, uint32_t b, uint32_t c) {
  uint32_t h = a * 0x9E3779B1u + b * 0x85EBCA77u + c * 0xC2B2AE3Du;
  h ^= h >> 16; h *= 0x7FEB352Du;
  h ^= h >> 15; h *= 0x846CA68Bu;
  h ^= h >> 16;
  return h;
}
__device__ __forceinline__ float u01(uint32_t h) {
  return (float)(h >> 8) * 0x1p-24f;
}

__global__ __launch_bounds__(BLOCK) void snn_lif_wmma(
    const float* __restrict__ x,
    const float* __restrict__ w1, const float* __restrict__ b1,
    const float* __restrict__ w2, const float* __restrict__ b2,
    float* __restrict__ out, int B)
{
  // Weights staged once, f16, column-major [col][k] so B-fragments are two
  // aligned ds_load_b128 per lane. Per-wave s1 scratch for the D->A transpose.
  __shared__ _Float16 w1lds[HPAD * KPAD];                 // 2 KB
  __shared__ _Float16 w2lds[OPAD * KPAD];                 // 13 KB
  __shared__ _Float16 s1lds[WAVES_PER_BLOCK][16 * KPAD];  // 4 KB

  const int tid = threadIdx.x;

  for (int i = tid; i < HPAD * KPAD; i += BLOCK) {
    int col = i >> 5, k = i & 31;
    float v = (col < HID && k < IN_F) ? w1[col * IN_F + k] : 0.0f;
    w1lds[i] = (_Float16)v;
  }
  for (int i = tid; i < OPAD * KPAD; i += BLOCK) {
    int col = i >> 5, k = i & 31;
    float v = (col < OUT_F && k < HID) ? w2[col * HID + k] : 0.0f;
    w2lds[i] = (_Float16)v;
  }
  {
    int w = tid >> 5, ln = tid & 31;
    #pragma unroll
    for (int e = 0; e < 16; ++e) s1lds[w][ln * 16 + e] = (_Float16)0.0f;
  }
  __syncthreads();

  const int wave = tid >> 5;
  const int lane = tid & 31;
  const int lh   = lane >> 4;      // which K-half of the wave this lane holds
  const int m    = lane & 15;      // A row-in-tile == B/C/D col-in-tile
  const int kb0  = lh ? 8 : 0;     // A-frag K bases (16-bit A layout)
  const int kb1  = lh ? 24 : 16;

  const int tile   = blockIdx.x * WAVES_PER_BLOCK + wave;
  const int nTiles = B >> 4;
  if (tile >= nTiles) return;
  const int rowBase = tile << 4;
  const int rowA    = rowBase + m; // batch row this lane's A slots cover

  // x probabilities for this lane's 16 A-fragment slots (pad k>=20 -> 0)
  float xv[16];
  #pragma unroll
  for (int e = 0; e < 16; ++e) {
    int k = (e < 8) ? (kb0 + e) : (kb1 + (e - 8));
    xv[e] = (k < IN_F) ? x[rowA * IN_F + k] : 0.0f;
  }

  // biases in C/D layout (all 8 accum regs in a lane share column N)
  float bias1[H_TILES];
  #pragma unroll
  for (int i = 0; i < H_TILES; ++i) {
    int col = i * 16 + m;
    bias1[i] = (col < HID) ? b1[col] : 0.0f;
  }
  float bias2[O_TILES];
  #pragma unroll
  for (int j = 0; j < O_TILES; ++j) {
    int col = j * 16 + m;
    bias2[j] = (col < OUT_F) ? b2[col] : 0.0f;
  }

  v8f mem1[H_TILES];
  v8f mem2[O_TILES];
  #pragma unroll
  for (int i = 0; i < H_TILES; ++i)
    #pragma unroll
    for (int e = 0; e < 8; ++e) mem1[i][e] = 0.0f;
  #pragma unroll
  for (int j = 0; j < O_TILES; ++j)
    #pragma unroll
    for (int e = 0; e < 8; ++e) mem2[j][e] = 0.0f;

  const _Float16 one = (_Float16)1.0f;
  const _Float16 zro = (_Float16)0.0f;
  _Float16* s1w = s1lds[wave];

  #pragma unroll 1
  for (int t = 0; t < NUM_STEPS; ++t) {
    // ---- Bernoulli input spikes directly in WMMA A-fragment layout ----
    v16h aSpk;
    #pragma unroll
    for (int e = 0; e < 16; ++e) {
      int k = (e < 8) ? (kb0 + e) : (kb1 + (e - 8));
      float u = u01(hash3((uint32_t)t, (uint32_t)rowA, (uint32_t)k));
      aSpk[e] = (u < xv[e]) ? one : zro;
    }

    // ---- layer 1: mem1 = beta*mem1 - s1_prev*thr + b1 + spikes@w1^T ----
    #pragma unroll
    for (int i = 0; i < H_TILES; ++i) {
      v8f c;
      #pragma unroll
      for (int e = 0; e < 8; ++e) {
        float mp = mem1[i][e];
        c[e] = BETA_H * mp - ((mp > THR_H) ? THR_H : 0.0f) + bias1[i];
      }
      const _Float16* wcol = &w1lds[(i * 16 + m) * KPAD];
      v8h blo = *(const v8h*)(wcol + kb0);
      v8h bhi = *(const v8h*)(wcol + kb1);
      v16h bf = __builtin_shufflevector(blo, bhi,
                  0,1,2,3,4,5,6,7,8,9,10,11,12,13,14,15);
      mem1[i] = __builtin_amdgcn_wmma_f32_16x16x32_f16(
                  false, aSpk, false, bf, (short)0, c, false, false);
    }

    // ---- s1 = H(mem1 - thr): D-layout -> row-major LDS (per-wave scratch) ----
    #pragma unroll
    for (int i = 0; i < H_TILES; ++i) {
      int col = i * 16 + m;
      if (col < HID) {
        #pragma unroll
        for (int e = 0; e < 8; ++e) {
          int row = e + lh * 8;
          s1w[row * KPAD + col] = (mem1[i][e] > THR_H) ? one : zro;
        }
      }
    }
    // same-wave LDS is processed in order; this pins compiler ordering
    asm volatile("s_wait_dscnt 0" ::: "memory");

    // ---- read s1 back as layer-2 A fragment (two aligned b128 per lane) ----
    const _Float16* arow = &s1w[m * KPAD];
    v8h alo = *(const v8h*)(arow + kb0);
    v8h ahi = *(const v8h*)(arow + kb1);
    v16h aS1 = __builtin_shufflevector(alo, ahi,
                 0,1,2,3,4,5,6,7,8,9,10,11,12,13,14,15);

    // ---- layer 2 + output spikes ----
    #pragma unroll
    for (int j = 0; j < O_TILES; ++j) {
      v8f c;
      #pragma unroll
      for (int e = 0; e < 8; ++e) {
        float mp = mem2[j][e];
        c[e] = BETA_O * mp - ((mp > THR_O) ? THR_O : 0.0f) + bias2[j];
      }
      const _Float16* wcol = &w2lds[(j * 16 + m) * KPAD];
      v8h blo = *(const v8h*)(wcol + kb0);
      v8h bhi = *(const v8h*)(wcol + kb1);
      v16h bf = __builtin_shufflevector(blo, bhi,
                  0,1,2,3,4,5,6,7,8,9,10,11,12,13,14,15);
      mem2[j] = __builtin_amdgcn_wmma_f32_16x16x32_f16(
                  false, aS1, false, bf, (short)0, c, false, false);

      int col = j * 16 + m;
      if (col < OUT_F) {
        float* op = out + (size_t)t * B * OUT_F + (size_t)rowBase * OUT_F + col;
        #pragma unroll
        for (int e = 0; e < 8; ++e) {
          int row = e + lh * 8;
          op[(size_t)row * OUT_F] = (mem2[j][e] > THR_O) ? 1.0f : 0.0f;
        }
      }
    }
  }
}

extern "C" void kernel_launch(void* const* d_in, const int* in_sizes, int n_in,
                              void* d_out, int out_size, void* d_ws, size_t ws_size,
                              hipStream_t stream) {
  (void)n_in; (void)out_size; (void)d_ws; (void)ws_size;
  const float* x  = (const float*)d_in[0];
  const float* w1 = (const float*)d_in[1];
  const float* b1 = (const float*)d_in[2];
  const float* w2 = (const float*)d_in[3];
  const float* b2 = (const float*)d_in[4];
  float* out = (float*)d_out;

  int B = in_sizes[0] / IN_F;                      // 32768
  int nTiles = (B + 15) / 16;                      // 16-row tiles, 1 per wave
  int blocks = (nTiles + WAVES_PER_BLOCK - 1) / WAVES_PER_BLOCK;
  snn_lif_wmma<<<blocks, BLOCK, 0, stream>>>(x, w1, b1, w2, b2, out, B);
}
